// AimNet2Core_52845277610672
// MI455X (gfx1250) — compile-verified, matrix operand order checked by reading, not run on hardware
//
#include <hip/hip_runtime.h>
#include <math.h>

#define NN   50000
#define PP   1600000

typedef __attribute__((ext_vector_type(16))) __bf16 v16bf;
typedef __attribute__((ext_vector_type(8)))  __bf16 v8bf;
typedef __attribute__((ext_vector_type(8)))  float  v8f;

// ---- workspace layout (bytes) ----
#define OFF_AGH   0u          // 65536 bf16 swizzled agh B (128x512)
#define OFF_WGF   131072u     // 4096  bf16 swizzled W_gf B (32x128, K padded)
#define OFF_W1    139264u     // 81920 bf16 swizzled W1 (320x256)
#define OFF_W2    303104u     // 32768 bf16 swizzled W2 (256x128)
#define OFF_W3    368640u     // 18432 bf16 swizzled W3 (128x144, cols>=130 zero)
#define OFF_S     405504u     // N*16 f32: S[n,g]  = sum_{p->n} gs[p,g]
#define OFF_V     3605504u    // N*48 f32: V[n,d,g]= sum_{p->n} gv[p,d,g]
// end: 13,205,504 B

// 16-bit WMMA A/B fragment element -> K index (documented CDNA5 layout)
__device__ __forceinline__ int frag_k(int i, int half) {
  return ((i >> 3) << 4) + (half << 3) + (i & 7);
}

__device__ __forceinline__ float gelu_exact(float x) {
  return 0.5f * x * (1.0f + erff(x * 0.70710678118654752440f));
}

// ------------------------------------------------------------------
__global__ void __launch_bounds__(256) zero_kernel(float* __restrict__ p, long n) {
  long i = (long)blockIdx.x * blockDim.x + threadIdx.x;
  long stride = (long)gridDim.x * blockDim.x;
  for (; i < n; i += stride) p[i] = 0.0f;
}

// ------------------------------------------------------------------
// bf16 B-fragment-swizzled copies of all weight matrices
__global__ void __launch_bounds__(256) swizzle_kernel(
    const float* __restrict__ agh, const float* __restrict__ wgf,
    const float* __restrict__ w1,  const float* __restrict__ w2,
    const float* __restrict__ w3,
    __bf16* __restrict__ agh_sw, __bf16* __restrict__ wgf_sw,
    __bf16* __restrict__ w1_sw,  __bf16* __restrict__ w2_sw,
    __bf16* __restrict__ w3_sw) {
  const int TOT_AGH = 65536, TOT_WGF = 4096, TOT_W1 = 81920, TOT_W2 = 32768, TOT_W3 = 18432;
  const int total = TOT_AGH + TOT_WGF + TOT_W1 + TOT_W2 + TOT_W3;
  for (int e0 = blockIdx.x * blockDim.x + threadIdx.x; e0 < total;
       e0 += gridDim.x * blockDim.x) {
    int e = e0;
    if (e < TOT_AGH) {  // agh: B = agh flattened (128 x 512), 32 ntiles x 4 ksteps
      int i = e & 15, L = (e >> 4) & 31, ks = (e >> 9) & 3, nt = e >> 11;
      int k = ks * 32 + frag_k(i, L >> 4);
      int n = nt * 16 + (L & 15);
      agh_sw[e] = (__bf16)agh[k * 512 + n];
      continue;
    }
    e -= TOT_AGH;
    if (e < TOT_WGF) {  // W_gf: 16x128, K padded to 32
      int i = e & 15, L = (e >> 4) & 31, nt = e >> 9;
      int k = frag_k(i, L >> 4);
      int col = nt * 16 + (L & 15);
      wgf_sw[e] = (k < 16) ? (__bf16)wgf[k * 128 + col] : (__bf16)0.0f;
      continue;
    }
    e -= TOT_WGF;
    if (e < TOT_W1) {  // W1: 320x256, 16 ntiles x 10 ksteps
      int i = e & 15, L = (e >> 4) & 31, t = e >> 9;
      int ks = t % 10, nt = t / 10;
      int k = ks * 32 + frag_k(i, L >> 4);
      w1_sw[e] = (__bf16)w1[k * 256 + nt * 16 + (L & 15)];
      continue;
    }
    e -= TOT_W1;
    if (e < TOT_W2) {  // W2: 256x128, 8 ntiles x 8 ksteps
      int i = e & 15, L = (e >> 4) & 31, t = e >> 9;
      int ks = t & 7, nt = t >> 3;
      int k = ks * 32 + frag_k(i, L >> 4);
      w2_sw[e] = (__bf16)w2[k * 128 + nt * 16 + (L & 15)];
      continue;
    }
    e -= TOT_W2;
    {  // W3: 128x130 padded to 144 cols, 9 ntiles x 4 ksteps
      int i = e & 15, L = (e >> 4) & 31, t = e >> 9;
      int ks = t & 3, nt = t >> 2;
      int k = ks * 32 + frag_k(i, L >> 4);
      int col = nt * 16 + (L & 15);
      w3_sw[e] = (col < 130) ? (__bf16)w3[k * 130 + col] : (__bf16)0.0f;
    }
  }
}

// ------------------------------------------------------------------
// Per-pair work reduced to its minimum: S[n] += gs[p], V[n] += gv[p].
// Fully coalesced reads; 16/48 consecutive lanes hit consecutive floats
// of the same destination row (atomics coalesce at L2).
__global__ void __launch_bounds__(256) scatter_kernel(
    const int* __restrict__ idxj, const float* __restrict__ gs,
    const float* __restrict__ gv, float* __restrict__ S,
    float* __restrict__ V) {
  const long t1 = (long)PP * 16;
  const long t2 = (long)PP * 48;
  long stride = (long)gridDim.x * blockDim.x;
  long i0 = (long)blockIdx.x * blockDim.x + threadIdx.x;
  for (long i = i0; i < t1; i += stride) {
    int p = (int)(i >> 4), g = (int)(i & 15);
    atomicAdd(S + (size_t)idxj[p] * 16 + g, gs[i]);
  }
  for (long i = i0; i < t2; i += stride) {
    int p = (int)(i / 48), c = (int)(i % 48);
    atomicAdd(V + (size_t)idxj[p] * 48 + c, gv[i]);
  }
}

// ------------------------------------------------------------------
// Node kernel: everything else, per wave-tile of 16 atoms, zero atomics.
//  mg = S@Wgf (WMMA)          -> msg[0:128]=a*mg, msg[160:288]=q*mg
//  T  = a@agh (WMMA, 128x512) -> avf[h,d] = sum_g V[d,g]*T[g,h] -> safe-norm
//  3-layer WMMA MLP with exact GELU -> route 130 cols to (delta_a,dq,f)
__global__ void __launch_bounds__(256) node_kernel(
    const float* __restrict__ emb, const float* __restrict__ charges,
    const float* __restrict__ Sarr, const float* __restrict__ Varr,
    const __bf16* __restrict__ wgf_sw, const __bf16* __restrict__ agh_sw,
    const __bf16* __restrict__ w1sw, const float* __restrict__ b1,
    const __bf16* __restrict__ w2sw, const float* __restrict__ b2,
    const __bf16* __restrict__ w3sw, const float* __restrict__ b3,
    float* __restrict__ out) {
  extern __shared__ char smem[];
  const int tid = threadIdx.x;
  const int wave = tid >> 5, lane = tid & 31;
  const int half = lane >> 4, lm = lane & 15;
  __bf16* AJb = (__bf16*)(smem) + wave * 2048;            // 16x128 bf16 a rows
  float*  Sf  = (float*)(smem + 32768) + wave * 256;      // 16x16 f32
  float*  Vf  = (float*)(smem + 40960) + wave * 768;      // 16x48 f32
  float*  Qf  = (float*)(smem + 65536) + wave * 16;       // 16 f32
  __bf16* MSG = (__bf16*)(smem + 66048) + wave * 5120;    // 16x320 bf16
  __bf16* H1s = (__bf16*)(smem + 147968) + wave * 4096;   // 16x256 bf16
  __bf16* H2s = (__bf16*)(smem + 213504) + wave * 2048;   // 16x128 bf16

  const int ntg = (NN + 127) / 128;
  for (int tg = blockIdx.x; tg < ntg; tg += gridDim.x) {
    const int abase = tg * 128 + wave * 16;
    // ---- stage a (bf16), S, V, q; zero the vec_q block of msg ----
    for (int t = lane; t < 16 * 32; t += 32) {
      int r = t >> 5, c4 = t & 31;
      int n = abase + r;
      float4 a = {0.f, 0.f, 0.f, 0.f};
      if (n < NN) a = ((const float4*)(emb + (size_t)n * 128))[c4];
      int c = c4 * 4;
      AJb[r * 128 + c + 0] = (__bf16)a.x;
      AJb[r * 128 + c + 1] = (__bf16)a.y;
      AJb[r * 128 + c + 2] = (__bf16)a.z;
      AJb[r * 128 + c + 3] = (__bf16)a.w;
    }
    for (int t = lane; t < 16 * 4; t += 32) {
      int r = t >> 2, c4 = t & 3;
      int n = abase + r;
      float4 s = {0.f, 0.f, 0.f, 0.f};
      if (n < NN) s = ((const float4*)(Sarr + (size_t)n * 16))[c4];
      ((float4*)Sf)[t] = s;
    }
    for (int t = lane; t < 16 * 12; t += 32) {
      int r = t / 12, c4 = t % 12;
      int n = abase + r;
      float4 vv = {0.f, 0.f, 0.f, 0.f};
      if (n < NN) vv = ((const float4*)(Varr + (size_t)n * 48))[c4];
      ((float4*)Vf)[t] = vv;
    }
    if (lane < 16) {
      int n = abase + lane;
      Qf[lane] = (n < NN) ? charges[n] : 0.f;
    }
    for (int t = lane; t < 16 * 32; t += 32) {
      int r = t >> 5, h = t & 31;
      MSG[r * 320 + 288 + h] = (__bf16)0.0f;  // vec_q = 0
    }
    __syncthreads();

    // ---- mg = S @ W_gf (K padded to 32); consume into msg cols 0:128/160:288
    {
      v16bf ga;
      const float* sp = Sf + lm * 16 + half * 8;
      float4 s0 = *(const float4*)sp;
      float4 s1 = *(const float4*)(sp + 4);
      ga[0] = (__bf16)s0.x; ga[1] = (__bf16)s0.y; ga[2] = (__bf16)s0.z; ga[3] = (__bf16)s0.w;
      ga[4] = (__bf16)s1.x; ga[5] = (__bf16)s1.y; ga[6] = (__bf16)s1.z; ga[7] = (__bf16)s1.w;
#pragma unroll
      for (int i = 8; i < 16; ++i) ga[i] = (__bf16)0.0f;
#pragma unroll
      for (int nt = 0; nt < 8; ++nt) {
        v16bf gb = *(const v16bf*)(wgf_sw + (size_t)(nt * 32 + lane) * 16);
        v8f c = {0.f, 0.f, 0.f, 0.f, 0.f, 0.f, 0.f, 0.f};
        c = __builtin_amdgcn_wmma_f32_16x16x32_bf16(false, ga, false, gb,
                                                    (short)0, c, false, false);
        int f = nt * 16 + lm;
#pragma unroll
        for (int v = 0; v < 8; ++v) {
          int m = v + half * 8;
          float a = (float)AJb[m * 128 + f];
          MSG[m * 320 + f]       = (__bf16)(a * c[v]);
          MSG[m * 320 + 160 + f] = (__bf16)(Qf[m] * c[v]);
        }
      }
    }

    // ---- T = a @ agh (16x128 @ 128x512); consume via V into avf; safe-norm
    {
      v16bf afr[4];
#pragma unroll
      for (int ks = 0; ks < 4; ++ks) {
        const __bf16* ap = AJb + lm * 128 + ks * 32 + half * 8;
        v8bf lo = *(const v8bf*)ap;
        v8bf hi = *(const v8bf*)(ap + 16);
#pragma unroll
        for (int i = 0; i < 8; ++i) { afr[ks][i] = lo[i]; afr[ks][8 + i] = hi[i]; }
      }
      float avf[8][2][3];
#pragma unroll
      for (int v = 0; v < 8; ++v)
#pragma unroll
        for (int hh = 0; hh < 2; ++hh)
#pragma unroll
          for (int d = 0; d < 3; ++d) avf[v][hh][d] = 0.f;

      for (int nt = 0; nt < 32; ++nt) {
        v8f acc = {0.f, 0.f, 0.f, 0.f, 0.f, 0.f, 0.f, 0.f};
#pragma unroll
        for (int ks = 0; ks < 4; ++ks) {
          v16bf bfr = *(const v16bf*)(agh_sw + (size_t)((nt * 4 + ks) * 32 + lane) * 16);
          acc = __builtin_amdgcn_wmma_f32_16x16x32_bf16(false, afr[ks], false, bfr,
                                                        (short)0, acc, false, false);
        }
        int g = nt >> 1, hh = nt & 1;  // column n = g*32 + hh*16 + lm
#pragma unroll
        for (int v = 0; v < 8; ++v) {
          int m = v + half * 8;
          float t = acc[v];
          avf[v][hh][0] += Vf[m * 48 + 0 * 16 + g] * t;
          avf[v][hh][1] += Vf[m * 48 + 1 * 16 + g] * t;
          avf[v][hh][2] += Vf[m * 48 + 2 * 16 + g] * t;
        }
      }
#pragma unroll
      for (int v = 0; v < 8; ++v) {
        int m = v + half * 8;
#pragma unroll
        for (int hh = 0; hh < 2; ++hh) {
          float sq = avf[v][hh][0] * avf[v][hh][0] +
                     avf[v][hh][1] * avf[v][hh][1] +
                     avf[v][hh][2] * avf[v][hh][2];
          float nv = (sq > 0.f) ? sqrtf(sq) : 0.f;  // safe_norm
          MSG[m * 320 + 128 + hh * 16 + lm] = (__bf16)nv;
        }
      }
    }
    __syncthreads();

    // ---- layer 1: 320 -> 256 (A-frags hoisted; depend only on k-step) ----
    {
      v16bf a1[10];
#pragma unroll
      for (int ks = 0; ks < 10; ++ks) {
        const __bf16* ap = MSG + lm * 320 + ks * 32 + half * 8;
        v8bf lo = *(const v8bf*)ap;
        v8bf hi = *(const v8bf*)(ap + 16);
#pragma unroll
        for (int i = 0; i < 8; ++i) { a1[ks][i] = lo[i]; a1[ks][8 + i] = hi[i]; }
      }
      for (int nt = 0; nt < 16; ++nt) {
        v8f acc = {0.f, 0.f, 0.f, 0.f, 0.f, 0.f, 0.f, 0.f};
#pragma unroll
        for (int ks = 0; ks < 10; ++ks) {
          v16bf bfr = *(const v16bf*)(w1sw + ((size_t)(nt * 10 + ks) * 32 + lane) * 16);
          acc = __builtin_amdgcn_wmma_f32_16x16x32_bf16(false, a1[ks], false, bfr,
                                                        (short)0, acc, false, false);
        }
        float bias = b1[nt * 16 + lm];
#pragma unroll
        for (int v = 0; v < 8; ++v) {
          int m = v + half * 8;
          H1s[m * 256 + nt * 16 + lm] = (__bf16)gelu_exact(acc[v] + bias);
        }
      }
    }
    __syncthreads();

    // ---- layer 2: 256 -> 128 ----
    {
      v16bf a2[8];
#pragma unroll
      for (int ks = 0; ks < 8; ++ks) {
        const __bf16* ap = H1s + lm * 256 + ks * 32 + half * 8;
        v8bf lo = *(const v8bf*)ap;
        v8bf hi = *(const v8bf*)(ap + 16);
#pragma unroll
        for (int i = 0; i < 8; ++i) { a2[ks][i] = lo[i]; a2[ks][8 + i] = hi[i]; }
      }
      for (int nt = 0; nt < 8; ++nt) {
        v8f acc = {0.f, 0.f, 0.f, 0.f, 0.f, 0.f, 0.f, 0.f};
#pragma unroll
        for (int ks = 0; ks < 8; ++ks) {
          v16bf bfr = *(const v16bf*)(w2sw + ((size_t)(nt * 8 + ks) * 32 + lane) * 16);
          acc = __builtin_amdgcn_wmma_f32_16x16x32_bf16(false, a2[ks], false, bfr,
                                                        (short)0, acc, false, false);
        }
        float bias = b2[nt * 16 + lm];
#pragma unroll
        for (int v = 0; v < 8; ++v) {
          int m = v + half * 8;
          H2s[m * 128 + nt * 16 + lm] = (__bf16)gelu_exact(acc[v] + bias);
        }
      }
    }
    __syncthreads();

    // ---- layer 3: 128 -> 130 (padded 144); route columns to outputs ----
    {
      v16bf a3[4];
#pragma unroll
      for (int ks = 0; ks < 4; ++ks) {
        const __bf16* ap = H2s + lm * 128 + ks * 32 + half * 8;
        v8bf lo = *(const v8bf*)ap;
        v8bf hi = *(const v8bf*)(ap + 16);
#pragma unroll
        for (int i = 0; i < 8; ++i) { a3[ks][i] = lo[i]; a3[ks][8 + i] = hi[i]; }
      }
      for (int nt = 0; nt < 9; ++nt) {
        v8f acc = {0.f, 0.f, 0.f, 0.f, 0.f, 0.f, 0.f, 0.f};
#pragma unroll
        for (int ks = 0; ks < 4; ++ks) {
          v16bf bfr = *(const v16bf*)(w3sw + ((size_t)(nt * 4 + ks) * 32 + lane) * 16);
          acc = __builtin_amdgcn_wmma_f32_16x16x32_bf16(false, a3[ks], false, bfr,
                                                        (short)0, acc, false, false);
        }
        int c = nt * 16 + lm;
        float bias = (c < 130) ? b3[c] : 0.f;
#pragma unroll
        for (int v = 0; v < 8; ++v) {
          int m = v + half * 8;
          int n = abase + m;
          float x = acc[v] + bias;
          if (n < NN && c < 130) {
            if (c == 0)
              out[(size_t)NN * 128 + n] = x;  // delta_q
            else if (c == 1)
              out[(size_t)NN * 128 + NN + n] = x;  // f
            else
              out[(size_t)n * 128 + (c - 2)] = x;  // delta_a
          }
        }
      }
    }
    __syncthreads();
  }
}

// ------------------------------------------------------------------
extern "C" void kernel_launch(void* const* d_in, const int* in_sizes, int n_in,
                              void* d_out, int out_size, void* d_ws,
                              size_t ws_size, hipStream_t stream) {
  const float* emb     = (const float*)d_in[0];
  const float* charges = (const float*)d_in[1];
  const int*   pidx    = (const int*)d_in[2];
  const float* gs      = (const float*)d_in[3];
  const float* gv      = (const float*)d_in[4];
  const float* agh     = (const float*)d_in[5];
  const float* wgf     = (const float*)d_in[6];
  const float* w1      = (const float*)d_in[7];
  const float* b1      = (const float*)d_in[8];
  const float* w2      = (const float*)d_in[9];
  const float* b2      = (const float*)d_in[10];
  const float* w3      = (const float*)d_in[11];
  const float* b3      = (const float*)d_in[12];
  float* out = (float*)d_out;

  char* ws = (char*)d_ws;
  __bf16* agh_sw = (__bf16*)(ws + OFF_AGH);
  __bf16* wgf_sw = (__bf16*)(ws + OFF_WGF);
  __bf16* w1_sw  = (__bf16*)(ws + OFF_W1);
  __bf16* w2_sw  = (__bf16*)(ws + OFF_W2);
  __bf16* w3_sw  = (__bf16*)(ws + OFF_W3);
  float* Sarr = (float*)(ws + OFF_S);
  float* Varr = (float*)(ws + OFF_V);

  (void)hipFuncSetAttribute((const void*)node_kernel,
                            hipFuncAttributeMaxDynamicSharedMemorySize, 246272);

  // S | V contiguous: zero as one flat region (N*64 floats)
  zero_kernel<<<1024, 256, 0, stream>>>(Sarr, (long)NN * 64);
  swizzle_kernel<<<512, 256, 0, stream>>>(agh, wgf, w1, w2, w3, agh_sw, wgf_sw,
                                          w1_sw, w2_sw, w3_sw);
  scatter_kernel<<<4096, 256, 0, stream>>>(pidx + PP, gs, gv, Sarr, Varr);
  node_kernel<<<391, 256, 246272, stream>>>(emb, charges, Sarr, Varr, wgf_sw,
                                            agh_sw, w1_sw, b1, w2_sw, b2, w3_sw,
                                            b3, out);
}
